// CustomLayer_87857851007262
// MI455X (gfx1250) — matrix operand order repaired and checked by reading, not compiled, not run on hardware
//
#include <hip/hip_runtime.h>

typedef __attribute__((ext_vector_type(16))) __bf16 v16bf;
typedef __attribute__((ext_vector_type(8)))  float  v8f;
typedef unsigned int u32;

#define TSZ   512
#define BATCH 1024
#define NFEAT 4096
#define G_MIN_C 1.0e-6f
#define G_MAX_C 1.0e-4f
#define RP_C    2.0f

// workspace byte offsets
#define WS_XSWZ 0ull                   // 8 MB  : x bf16, A-fragment swizzled
#define WS_GSWZ (8ull  << 20)          // 32 MB : g_eff bf16, B-fragment swizzled
#define WS_CSWZ (40ull << 20)          // 32 MB : cond  bf16, B-fragment swizzled
#define WS_SARR (72ull << 20)          // 64 floats : per-tile scale s
#define WS_WMIN ((72ull << 20) + 256)  // 64 floats : per-tile wmin
#define WS_RSUM ((72ull << 20) + 512)  // 8192 floats : rowsum[itile][b]

__device__ __forceinline__ u32 bf16pk(float a, float b) {
  // round-to-nearest-even f32 -> bf16, packed pair
  u32 ua = __float_as_uint(a), ub = __float_as_uint(b);
  ua = (ua + 0x7FFFu + ((ua >> 16) & 1u)) >> 16;
  ub = (ub + 0x7FFFu + ((ub >> 16) & 1u)) >> 16;
  return (ua & 0xFFFFu) | (ub << 16);
}

union Frag { uint4 q[2]; v16bf v; };

// ---------------------------------------------------------------- K1: per-tile min/max
__global__ void k_tile_minmax(const float* __restrict__ w,
                              float* __restrict__ s_arr,
                              float* __restrict__ wmin_arr) {
  __shared__ float smn[256], smx[256];
  int tile = blockIdx.x;                 // 0..63  (it*8 + ot)
  int it = tile >> 3, ot = tile & 7;
  const float* base = w + (size_t)it * TSZ * NFEAT + (size_t)ot * TSZ;
  float mn = 3.4e38f, mx = -3.4e38f;
  for (int idx = threadIdx.x; idx < TSZ * TSZ; idx += 256) {
    int r = idx >> 9, c = idx & 511;
    float v = base[(size_t)r * NFEAT + c];
    mn = fminf(mn, v); mx = fmaxf(mx, v);
  }
  smn[threadIdx.x] = mn; smx[threadIdx.x] = mx;
  __syncthreads();
  for (int off = 128; off > 0; off >>= 1) {
    if ((int)threadIdx.x < off) {
      smn[threadIdx.x] = fminf(smn[threadIdx.x], smn[threadIdx.x + off]);
      smx[threadIdx.x] = fmaxf(smx[threadIdx.x], smx[threadIdx.x + off]);
    }
    __syncthreads();
  }
  if (threadIdx.x == 0) {
    float wmin = smn[0], wmax = smx[0];
    s_arr[tile]   = (G_MAX_C - G_MIN_C) / (wmax - wmin + 1e-12f);
    wmin_arr[tile] = wmin;
  }
}

// ---------------------------------------------------------------- K2: x row-tile sums
__global__ void k_rowsum(const float* __restrict__ x, float* __restrict__ rsum) {
  int gid = blockIdx.x * 256 + threadIdx.x;   // 8192 = 8 itiles * 1024 rows
  int it = gid >> 10, b = gid & 1023;
  const float* p = x + (size_t)b * NFEAT + it * TSZ;
  float s = 0.f;
  for (int k = 0; k < TSZ; ++k) s += p[k];
  rsum[gid] = s;
}

// ---------------------------------------------------------------- K3: x -> bf16 A-fragment swizzle
// ISA A layout (16-bit 16x32): row m = lane&15; lanes<16: vgpr0-3 K=0..7, vgpr4-7 K=16..23;
// lanes>=16: vgpr0-3 K=8..15, vgpr4-7 K=24..31; each vgpr = 2 consecutive K.
__global__ void k_xswz(const float* __restrict__ x, u32* __restrict__ xs) {
  int wave = threadIdx.x >> 5, lane = threadIdx.x & 31;
  int gw  = blockIdx.x * 8 + wave;            // 0..8191 = it(8) * rowTile(64) * kt(16)
  int it  = gw >> 10;
  int rem = gw & 1023;
  int rowTile = rem >> 4, kt = rem & 15;
  int m = lane & 15, grp = lane >> 4;
  int b = rowTile * 16 + m;
  const float* xr = x + (size_t)b * NFEAT + it * TSZ + kt * 32;
  u32* dst = xs + ((((size_t)it * 64 + rowTile) * 16 + kt) * 256) + lane * 8;
#pragma unroll
  for (int p = 0; p < 8; ++p) {
    int k0 = 2 * (p & 3) + ((p >= 4) ? 16 : 0) + grp * 8;
    dst[p] = bf16pk(xr[k0], xr[k0 + 1]);
  }
}

// ---------------------------------------------------------------- K4: weight -> cond/g_eff, B-fragment swizzle
// B layout (16-bit 32x16), matching A: col n = lane&15; lanes<16: vgpr0-3 K=0..7, vgpr4-7 K=16..23; lanes>=16: vgpr0-3 K=8..15, vgpr4-7 K=24..31.
__global__ void k_wswz(const float* __restrict__ w,
                       const float* __restrict__ s_arr,
                       const float* __restrict__ wmin_arr,
                       u32* __restrict__ gs, u32* __restrict__ cs) {
  int wave = threadIdx.x >> 5, lane = threadIdx.x & 31;
  int gw = blockIdx.x * 8 + wave;             // 0..32767 = tile(64) * kt(16) * nt(32)
  int tile = gw >> 9;
  int blk  = gw & 511;
  int kt = blk >> 5, nt = blk & 31;
  int it = tile >> 3, ot = tile & 7;
  float s = s_arr[tile], wmin = wmin_arr[tile];
  const float step  = (G_MAX_C - G_MIN_C) / 15.0f;   // 2**4 - 1 levels
  const float istep = 1.0f / step;
  int ct = nt * 16 + (lane & 15);             // col within tile (j)
  int cg = ot * TSZ + ct;                     // global col
  int grp = lane >> 4;
  size_t base = ((((size_t)tile) * 16 + kt) * 32 + nt) * 256 + lane * 8;
#pragma unroll
  for (int p = 0; p < 8; ++p) {
    int r0 = kt * 32 + 2 * (p & 3) + ((p >= 4) ? 16 : 0) + grp * 8;      // within-tile row (i) for h=0
    float gv[2], cv[2];
#pragma unroll
    for (int h = 0; h < 2; ++h) {
      int rt = r0 + h;
      float wv = w[((size_t)(it * TSZ + rt)) * NFEAT + cg];
      float cond = (wv - wmin) * s + G_MIN_C;
      cond = rintf((cond - G_MIN_C) * istep) * step + G_MIN_C;   // 4-bit quantization
      float rw = RP_C * ((float)(TSZ - rt) + (float)(ct + 1));   // Jeong wire resistance
      gv[h] = 1.0f / (1.0f / cond + rw);
      cv[h] = cond;
    }
    gs[base + p] = bf16pk(gv[0], gv[1]);
    cs[base + p] = bf16pk(cv[0], cv[1]);
  }
}

// ---------------------------------------------------------------- K5: main crossbar GEMM + correction
// Grid: (otile 8, rowTile 64); 256 threads = 8 waves; wave owns 4 column tiles (64 cols).
__global__ void __launch_bounds__(256)
k_xbar_gemm(const u32* __restrict__ xs, const u32* __restrict__ gs,
            const u32* __restrict__ cs, const float* __restrict__ s_arr,
            const float* __restrict__ wmin_arr, const float* __restrict__ rsum,
            const float* __restrict__ bias, float* __restrict__ out) {
  __shared__ float statP[8][16][6];   // per-wave partial row stats
  __shared__ float statF[16][4];      // final per-row {coeff, meanC, meanI, offrow}
  int ot = blockIdx.x;
  int rowTile = blockIdx.y;
  int wave = threadIdx.x >> 5, lane = threadIdx.x & 31;
  int grp = lane >> 4, n15 = lane & 15;

  v8f outAcc[4];
#pragma unroll
  for (int t = 0; t < 4; ++t) outAcc[t] = {};

  for (int itl = 0; itl < 8; ++itl) {
    float s    = s_arr[itl * 8 + ot];
    float wmin = wmin_arr[itl * 8 + ot];
    float inv_s = 1.0f / s;
    float offc  = G_MIN_C - s * wmin;

    v8f accC[4], accI[4];
#pragma unroll
    for (int t = 0; t < 4; ++t) { accC[t] = {}; accI[t] = {}; }

    const u32* xbase = xs + (((size_t)itl * 64 + rowTile) * 16) * 256;
    size_t tbase = ((size_t)(itl * 8 + ot)) * 16 * 32 * 256;

    for (int kt = 0; kt < 16; ++kt) {
      Frag a;
      const uint4* ap = (const uint4*)(xbase + (size_t)kt * 256 + lane * 8);
      a.q[0] = ap[0]; a.q[1] = ap[1];
#pragma unroll
      for (int t = 0; t < 4; ++t) {
        int nt = wave * 4 + t;
        size_t boff = tbase + ((size_t)kt * 32 + nt) * 256 + lane * 8;
        Frag bg, bc;
        const uint4* gp = (const uint4*)(gs + boff);
        const uint4* cp = (const uint4*)(cs + boff);
        bg.q[0] = gp[0]; bg.q[1] = gp[1];
        bc.q[0] = cp[0]; bc.q[1] = cp[1];
        accC[t] = __builtin_amdgcn_wmma_f32_16x16x32_bf16(
            false, a.v, false, bg.v, (short)0, accC[t], false, false);
        accI[t] = __builtin_amdgcn_wmma_f32_16x16x32_bf16(
            false, a.v, false, bc.v, (short)0, accI[t], false, false);
      }
    }

    // --- row stats: C/D layout => lane holds col n15, vgpr v holds row v + 8*grp
#pragma unroll
    for (int v = 0; v < 8; ++v) {
      float mnC = accC[0][v], mxC = mnC, smC = 0.f;
      float mnI = accI[0][v], mxI = mnI, smI = 0.f;
#pragma unroll
      for (int t = 0; t < 4; ++t) {
        float c = accC[t][v], i = accI[t][v];
        mnC = fminf(mnC, c); mxC = fmaxf(mxC, c); smC += c;
        mnI = fminf(mnI, i); mxI = fmaxf(mxI, i); smI += i;
      }
#pragma unroll
      for (int msk = 1; msk < 16; msk <<= 1) {   // reduce across the 16-lane row group
        mnC = fminf(mnC, __shfl_xor(mnC, msk, 32));
        mxC = fmaxf(mxC, __shfl_xor(mxC, msk, 32));
        smC += __shfl_xor(smC, msk, 32);
        mnI = fminf(mnI, __shfl_xor(mnI, msk, 32));
        mxI = fmaxf(mxI, __shfl_xor(mxI, msk, 32));
        smI += __shfl_xor(smI, msk, 32);
      }
      if (n15 == 0) {
        int row = v + grp * 8;
        statP[wave][row][0] = mnC; statP[wave][row][1] = mxC; statP[wave][row][2] = smC;
        statP[wave][row][3] = mnI; statP[wave][row][4] = mxI; statP[wave][row][5] = smI;
      }
    }
    __syncthreads();
    if (wave == 0 && lane < 16) {
      int row = lane;
      float mnC = statP[0][row][0], mxC = statP[0][row][1], smC = statP[0][row][2];
      float mnI = statP[0][row][3], mxI = statP[0][row][4], smI = statP[0][row][5];
#pragma unroll
      for (int w8 = 1; w8 < 8; ++w8) {
        mnC = fminf(mnC, statP[w8][row][0]); mxC = fmaxf(mxC, statP[w8][row][1]); smC += statP[w8][row][2];
        mnI = fminf(mnI, statP[w8][row][3]); mxI = fmaxf(mxI, statP[w8][row][4]); smI += statP[w8][row][5];
      }
      float coeff  = (mxI - mnI) / (mxC - mnC + 1e-8f);
      float meanC  = smC * (1.0f / 512.0f);
      float meanI  = smI * (1.0f / 512.0f);
      float offrow = rsum[itl * 1024 + rowTile * 16 + row] * offc;
      statF[row][0] = coeff; statF[row][1] = meanC; statF[row][2] = meanI; statF[row][3] = offrow;
    }
    __syncthreads();
#pragma unroll
    for (int v = 0; v < 8; ++v) {
      int row = v + grp * 8;
      float coeff = statF[row][0], meanC = statF[row][1];
      float meanI = statF[row][2], offrow = statF[row][3];
#pragma unroll
      for (int t = 0; t < 4; ++t) {
        float c = accC[t][v];
        outAcc[t][v] += ((c - meanC) * coeff + meanI - offrow) * inv_s;
      }
    }
  }

  // --- store with bias
#pragma unroll
  for (int t = 0; t < 4; ++t) {
    int col = ot * TSZ + wave * 64 + t * 16 + n15;
    float bv = bias[col];
#pragma unroll
    for (int v = 0; v < 8; ++v) {
      int b = rowTile * 16 + v + grp * 8;
      out[(size_t)b * NFEAT + col] = outAcc[t][v] + bv;
    }
  }
}

// ----------------------------------------------------------------
extern "C" void kernel_launch(void* const* d_in, const int* in_sizes, int n_in,
                              void* d_out, int out_size, void* d_ws, size_t ws_size,
                              hipStream_t stream) {
  const float* x    = (const float*)d_in[0];
  const float* w    = (const float*)d_in[1];
  const float* bias = (const float*)d_in[2];
  float* out = (float*)d_out;
  char* ws = (char*)d_ws;

  u32*   xs       = (u32*)(ws + WS_XSWZ);
  u32*   gs       = (u32*)(ws + WS_GSWZ);
  u32*   cs       = (u32*)(ws + WS_CSWZ);
  float* s_arr    = (float*)(ws + WS_SARR);
  float* wmin_arr = (float*)(ws + WS_WMIN);
  float* rsum     = (float*)(ws + WS_RSUM);

  k_tile_minmax<<<64,   256, 0, stream>>>(w, s_arr, wmin_arr);
  k_rowsum    <<<32,    256, 0, stream>>>(x, rsum);
  k_xswz      <<<1024,  256, 0, stream>>>(x, xs);
  k_wswz      <<<4096,  256, 0, stream>>>(w, s_arr, wmin_arr, gs, cs);
  k_xbar_gemm <<<dim3(8, 64), 256, 0, stream>>>(xs, gs, cs, s_arr, wmin_arr, rsum, bias, out);
}
